// ControlledResNet_300647711162
// MI455X (gfx1250) — matrix operand order repaired and checked by reading, not compiled, not run on hardware
//
#include <hip/hip_runtime.h>
#include <hip/hip_bf16.h>

// ---------------- problem constants (from reference) ----------------
#define NN 512
#define TT 128
#define PP 16
#define LL 128
#define HH 512
#define NLAYERS 2

// ---------------- tiling ----------------
#define ROWS    16                 // rows (samples) per block = one WMMA M tile
#define THREADS 256                // 8 waves of 32 (2 waves/SIMD -> big VGPR budget)
#define WAVES   (THREADS / 32)
#define CT_H    (HH / 16 / WAVES)      // 4 column tiles per wave in H layers
#define CT_O    (LL * PP / 16 / WAVES) // 16 column tiles per wave in head
#define ZSTR    (LL + 8)           // padded LDS stride (halves) for z  (136)
#define HSTR    (HH + 8)           // padded LDS stride (halves) for h  (520)

typedef __attribute__((ext_vector_type(16))) _Float16 v16h;
typedef __attribute__((ext_vector_type(8)))  _Float16 v8h;
typedef __attribute__((ext_vector_type(8)))  float    v8f;

// Branchless native tanh: tanh(x) = sign(x) * (1 - 2/(exp(2|x|)+1))
__device__ __forceinline__ float fast_tanh(float x) {
  const float TWO_LOG2E = 2.885390081777926f;          // 2*log2(e)
  float ax = __builtin_fabsf(x);
  float e  = __builtin_amdgcn_exp2f(ax * TWO_LOG2E);   // exp(2|x|)
  float r  = __builtin_fmaf(__builtin_amdgcn_rcpf(e + 1.0f), -2.0f, 1.0f);
  return __builtin_copysignf(r, x);
}

// A-fragment (16x32 f16) from LDS, row-major [m][k], stride `str` halves.
// Lanes 0-15 -> M=lane, K {0..7,16..23}; lanes 16-31 -> M=lane-16, K {8..15,24..31}.
__device__ __forceinline__ v16h load_a_frag(const _Float16* src, int lane,
                                            int str, int kt) {
  const int m  = lane & 15;
  const int kh = lane >> 4;
  const int k1 = kt * 32 + 8 * kh;
  const _Float16* p = src + m * str;
  v8h lo = *(const v8h*)(p + k1);
  v8h hi = *(const v8h*)(p + k1 + 16);
  v16h a;
#pragma unroll
  for (int i = 0; i < 8; ++i) { a[i] = lo[i]; a[i + 8] = hi[i]; }
  return a;
}

// B-fragment (32x16 f16) from row-major KxN f16 weights; N = lane&15,
// lanes 0-15 -> K {0..7,16..23}; lanes 16-31 -> K {8..15,24..31} (strided gather).
__device__ __forceinline__ v16h load_b_frag(const _Float16* W, int lane,
                                            int ntot, int k0, int n0) {
  const int kh = lane >> 4;
  const _Float16* p = W + (size_t)(k0 + 8 * kh) * ntot + n0 + (lane & 15);
  v16h b;
#pragma unroll
  for (int i = 0; i < 8; ++i) { b[i] = p[(size_t)i * ntot]; b[i + 8] = p[(size_t)(i + 16) * ntot]; }
  return b;
}

// ---------------- f32 -> f16 weight conversion ----------------
__global__ void cres_cvt_f16(const float* __restrict__ src,
                             unsigned short* __restrict__ dst, int n) {
  int i = blockIdx.x * blockDim.x + threadIdx.x;
  if (i < n) {
    _Float16 h = (_Float16)src[i];
    dst[i] = __builtin_bit_cast(unsigned short, h);
  }
}

// ------------------------------------------------------------------
// Persistent per-block scan: block b owns rows [16b, 16b+16).
// K-outer / accumulator-resident GEMM phases: each A fragment is loaded
// from LDS once per K tile and feeds all of this wave's column tiles.
// ------------------------------------------------------------------
__global__ __launch_bounds__(THREADS, 1)
void cres_scan_kernel(const float* __restrict__ path,
                      const float* __restrict__ phi_w,
                      const float* __restrict__ phi_b,
                      const float* __restrict__ alpha_w,
                      const float* __restrict__ alpha_b,
                      const float* __restrict__ b_in,
                      const float* __restrict__ bs,
                      const float* __restrict__ b_out,
                      const unsigned short* __restrict__ Win16u,
                      const unsigned short* __restrict__ Ws16u,
                      const unsigned short* __restrict__ Wout16u,
                      float* __restrict__ out) {
  const _Float16* Win16  = reinterpret_cast<const _Float16*>(Win16u);
  const _Float16* Ws16   = reinterpret_cast<const _Float16*>(Ws16u);
  const _Float16* Wout16 = reinterpret_cast<const _Float16*>(Wout16u);

  __shared__ float     z32[ROWS * LL];        // 8 KB   current state f32
  __shared__ _Float16  z16[ROWS * ZSTR];      // 4.25KB A-operand copy of z
  __shared__ _Float16  h16[2][ROWS * HSTR];   // 32.5KB double-buffered h (f16)
  __shared__ float     dxl[ROWS * PP];        // 1 KB   path increment tile
  __shared__ float     awl[LL];               // 0.5KB  alpha_w staged

  const int tid  = threadIdx.x;
  const int lane = tid & 31;
  const int wave = tid >> 5;
  const int kh   = lane >> 4;   // half-wave: selects M offset 0/8
  const int nl   = lane & 15;   // N within tile
  const int row0 = blockIdx.x * ROWS;

  if (tid < LL) awl[tid] = alpha_w[tid];

  // ---- z0 = path[:,0,:] @ phi_w + phi_b ----
  for (int idx = tid; idx < ROWS * LL; idx += THREADS) {
    const int m = idx >> 7, lc = idx & (LL - 1);
    const float* prow = path + (size_t)(row0 + m) * TT * PP;
    float s = phi_b[lc];
#pragma unroll
    for (int p = 0; p < PP; ++p) s += prow[p] * phi_w[p * LL + lc];
    z32[m * LL + lc] = s;
  }
  __syncthreads();

  // ---- intensity at t=0 (wave covers rows 2w, 2w+1; half-wave per row) ----
  {
    const int m = wave * 2 + kh;
    float s = 0.f;
#pragma unroll
    for (int j = 0; j < LL / 16; ++j)
      s += z32[m * LL + nl + 16 * j] * awl[nl + 16 * j];
#pragma unroll
    for (int off = 1; off < 16; off <<= 1) s += __shfl_xor(s, off, 32);
    if (nl == 0) out[(size_t)(row0 + m) * TT + 0] = s + alpha_b[0];
  }

  v8f resid[CT_H];  // per-lane f32 residual for this wave's column tiles

  for (int t = 0; t < TT - 1; ++t) {
    // ---- stage dx tile and f16 copy of z ----
    {
      const int m = tid >> 4, p = tid & 15;  // THREADS == ROWS*PP
      const float* prow = path + (size_t)(row0 + m) * TT * PP;
      dxl[tid] = prow[(t + 1) * PP + p] - prow[t * PP + p];
    }
    for (int idx = tid; idx < ROWS * LL; idx += THREADS) {
      const int m = idx >> 7, lc = idx & (LL - 1);
      z16[m * ZSTR + lc] = (_Float16)z32[idx];
    }
    __syncthreads();

    // ---- input layer: h = tanh(z @ W_in + b_in)  (M16,K128,N512) ----
    {
      v8f acc[CT_H];
#pragma unroll
      for (int j = 0; j < CT_H; ++j) {
        const float bv = b_in[(wave * CT_H + j) * 16 + nl];
#pragma unroll
        for (int r = 0; r < 8; ++r) acc[j][r] = bv;
      }
#pragma unroll 2
      for (int kt = 0; kt < LL / 32; ++kt) {
        v16h a = load_a_frag(z16, lane, ZSTR, kt);
#pragma unroll
        for (int j = 0; j < CT_H; ++j) {
          v16h b = load_b_frag(Win16, lane, HH, kt * 32, (wave * CT_H + j) * 16);
          acc[j] = __builtin_amdgcn_wmma_f32_16x16x32_f16(false, a, false, b,
                                                          (short)0, acc[j], false, false);
        }
      }
#pragma unroll
      for (int j = 0; j < CT_H; ++j) {
        const int n0 = (wave * CT_H + j) * 16;
#pragma unroll
        for (int r = 0; r < 8; ++r) {
          const float v = fast_tanh(acc[j][r]);
          resid[j][r] = v;
          h16[0][(r + 8 * kh) * HSTR + n0 + nl] = (_Float16)v;
        }
      }
    }
    __syncthreads();

    // ---- residual layers: h += tanh(h @ Ws[i] + bs[i])  (M16,K512,N512) ----
    int cur = 0;
    for (int li = 0; li < NLAYERS; ++li) {
      const int nxt = cur ^ 1;
      const _Float16* W = Ws16 + (size_t)li * HH * HH;
      const float* bsl = bs + li * HH;
      v8f acc[CT_H];
#pragma unroll
      for (int j = 0; j < CT_H; ++j) {
        const float bv = bsl[(wave * CT_H + j) * 16 + nl];
#pragma unroll
        for (int r = 0; r < 8; ++r) acc[j][r] = bv;
      }
#pragma unroll 2
      for (int kt = 0; kt < HH / 32; ++kt) {
        v16h a = load_a_frag(h16[cur], lane, HSTR, kt);
#pragma unroll
        for (int j = 0; j < CT_H; ++j) {
          v16h b = load_b_frag(W, lane, HH, kt * 32, (wave * CT_H + j) * 16);
          acc[j] = __builtin_amdgcn_wmma_f32_16x16x32_f16(false, a, false, b,
                                                          (short)0, acc[j], false, false);
        }
      }
#pragma unroll
      for (int j = 0; j < CT_H; ++j) {
        const int n0 = (wave * CT_H + j) * 16;
#pragma unroll
        for (int r = 0; r < 8; ++r) {
          const float v = resid[j][r] + fast_tanh(acc[j][r]);
          resid[j][r] = v;
          h16[nxt][(r + 8 * kh) * HSTR + n0 + nl] = (_Float16)v;
        }
      }
      cur = nxt;
      __syncthreads();
    }

    // ---- head + contraction: z += einsum('lp,p->l', F, dx) ----
    // One 16-col tile == one l index; N-within-tile == p.  (M16,K512,N2048)
    {
      v8f acc[CT_O];   // 16 x v8f = 128 VGPRs, resident across the K loop
#pragma unroll
      for (int j = 0; j < CT_O; ++j) {
        const float bv = b_out[(wave * CT_O + j) * 16 + nl];
#pragma unroll
        for (int r = 0; r < 8; ++r) acc[j][r] = bv;
      }
#pragma unroll 2
      for (int kt = 0; kt < HH / 32; ++kt) {
        v16h a = load_a_frag(h16[cur], lane, HSTR, kt);
#pragma unroll
        for (int j = 0; j < CT_O; ++j) {
          v16h b = load_b_frag(Wout16, lane, LL * PP, kt * 32, (wave * CT_O + j) * 16);
          acc[j] = __builtin_amdgcn_wmma_f32_16x16x32_f16(false, a, false, b,
                                                          (short)0, acc[j], false, false);
        }
      }
#pragma unroll
      for (int j = 0; j < CT_O; ++j) {
        const int ct = wave * CT_O + j;   // l index
#pragma unroll
        for (int r = 0; r < 8; ++r) {
          const int m = r + 8 * kh;
          float v = acc[j][r] * dxl[m * PP + nl];
#pragma unroll
          for (int off = 1; off < 16; off <<= 1) v += __shfl_xor(v, off, 32);
          if (nl == 0) z32[m * LL + ct] += v;
        }
      }
    }
    __syncthreads();

    // ---- intensity at t+1 ----
    {
      const int m = wave * 2 + kh;
      float s = 0.f;
#pragma unroll
      for (int j2 = 0; j2 < LL / 16; ++j2)
        s += z32[m * LL + nl + 16 * j2] * awl[nl + 16 * j2];
#pragma unroll
      for (int off = 1; off < 16; off <<= 1) s += __shfl_xor(s, off, 32);
      if (nl == 0) out[(size_t)(row0 + m) * TT + (t + 1)] = s + alpha_b[0];
    }
    __syncthreads();
  }
}

// ------------------------------------------------------------------
extern "C" void kernel_launch(void* const* d_in, const int* in_sizes, int n_in,
                              void* d_out, int out_size, void* d_ws, size_t ws_size,
                              hipStream_t stream) {
  const float* path    = (const float*)d_in[0];
  const float* phi_w   = (const float*)d_in[1];
  const float* phi_b   = (const float*)d_in[2];
  const float* alpha_w = (const float*)d_in[3];
  const float* alpha_b = (const float*)d_in[4];
  const float* W_in    = (const float*)d_in[5];
  const float* b_in    = (const float*)d_in[6];
  const float* Ws      = (const float*)d_in[7];
  const float* bs      = (const float*)d_in[8];
  const float* W_out   = (const float*)d_in[9];
  const float* b_out   = (const float*)d_in[10];

  // f16 weight copies in scratch (3.3 MB -> resident in the 192 MB L2)
  const int nWin  = LL * HH;            // 65536
  const int nWs   = NLAYERS * HH * HH;  // 524288
  const int nWout = HH * LL * PP;       // 1048576
  unsigned short* Win16  = (unsigned short*)d_ws;
  unsigned short* Ws16   = Win16 + nWin;
  unsigned short* Wout16 = Ws16 + nWs;

  cres_cvt_f16<<<(nWin  + 255) / 256, 256, 0, stream>>>(W_in,  Win16,  nWin);
  cres_cvt_f16<<<(nWs   + 255) / 256, 256, 0, stream>>>(Ws,    Ws16,   nWs);
  cres_cvt_f16<<<(nWout + 255) / 256, 256, 0, stream>>>(W_out, Wout16, nWout);

  cres_scan_kernel<<<NN / ROWS, THREADS, 0, stream>>>(
      path, phi_w, phi_b, alpha_w, alpha_b, b_in, bs, b_out,
      Win16, Ws16, Wout16, (float*)d_out);
}
